// Encoder_3942779977787
// MI455X (gfx1250) — matrix operand order repaired and checked by reading, not compiled, not run on hardware
//
#include <hip/hip_runtime.h>
#include <hip/hip_bf16.h>
#include <math.h>

// ---------------------------------------------------------------------------
// CDNA5 (gfx1250) transformer encoder: all matmuls via v_wmma_f32_16x16x32_bf16
// ---------------------------------------------------------------------------

typedef __bf16 bf16;
typedef __attribute__((ext_vector_type(16))) __bf16 v16bf;
typedef __attribute__((ext_vector_type(8)))  __bf16 v8bf;
typedef __attribute__((ext_vector_type(8)))  float  v8f;

#define K_DIM   1024
#define K_SEQ   2048
#define K_BATCH 2
#define K_ROWS  4096      // batch * seq
#define K_HEADS 16
#define K_DHEAD 64
#define K_FF    4096
#define K_QKVN  3072      // 3 * dim

// ---------------- WMMA helpers (layouts per CDNA5 ISA 7.12.2) ---------------

__device__ __forceinline__ v8f wmma_bf16(v16bf a, v16bf b, v8f c) {
  // D(16x16,f32) = A(16x32,bf16) * B(32x16,bf16) + C
  return __builtin_amdgcn_wmma_f32_16x16x32_bf16(
      /*neg_a=*/false, a, /*neg_b=*/false, b,
      /*c_mod=*/(short)0, c, /*reuse_a=*/false, /*reuse_b=*/false);
}

// A fragment: row m = m0 + lane%16; lo half lanes k = k0+0..7, hi half k0+8..15,
// second 8 elements at +16 in K.  Two contiguous 16-byte loads per lane.
__device__ __forceinline__ v16bf load_frag_a(const bf16* base, int ld, int m0, int k0) {
  int lane = threadIdx.x & 31;
  const bf16* p = base + (size_t)(m0 + (lane & 15)) * ld + (k0 + ((lane >> 4) << 3));
  v8bf lo = *(const v8bf*)p;
  v8bf hi = *(const v8bf*)(p + 16);
  return __builtin_shufflevector(lo, hi, 0,1,2,3,4,5,6,7,8,9,10,11,12,13,14,15);
}

// B fragment from a [N][K] (transposed) array: col n = n0 + lane%16;
// lo half lanes k = k0+0..7, hi half k0+8..15, second 8 elements at +16 in K.
__device__ __forceinline__ v16bf load_frag_b(const bf16* baseT, int ld, int k0, int n0) {
  int lane = threadIdx.x & 31;
  const bf16* p = baseT + (size_t)(n0 + (lane & 15)) * ld + (k0 + ((lane >> 4) << 3));
  v8bf lo = *(const v8bf*)p;
  v8bf hi = *(const v8bf*)(p + 16);
  return __builtin_shufflevector(lo, hi, 0,1,2,3,4,5,6,7,8,9,10,11,12,13,14,15);
}

// reductions within 16-lane halves (rows live per-half in the C layout)
__device__ __forceinline__ float half16_max(float x) {
  x = fmaxf(x, __shfl_xor(x, 1, 16));
  x = fmaxf(x, __shfl_xor(x, 2, 16));
  x = fmaxf(x, __shfl_xor(x, 4, 16));
  x = fmaxf(x, __shfl_xor(x, 8, 16));
  return x;
}
__device__ __forceinline__ float half16_sum(float x) {
  x += __shfl_xor(x, 1, 16);
  x += __shfl_xor(x, 2, 16);
  x += __shfl_xor(x, 4, 16);
  x += __shfl_xor(x, 8, 16);
  return x;
}

// ---------------- weight repack: f32 [K][N] -> bf16 [N][K] ------------------

__global__ void __launch_bounds__(256) transpose_w_k(const float* __restrict__ W,
                                                     bf16* __restrict__ WT,
                                                     int K, int N) {
  long i = (long)blockIdx.x * 256 + threadIdx.x;
  if (i >= (long)K * N) return;
  int n = (int)(i / K);
  int k = (int)(i % K);
  WT[i] = (bf16)W[(long)k * N + n];
}

// V transpose: qkv[b*2048+j][2048 + h*64 + d] -> vT[((b*16+h)*64+d)*2048 + j]
__global__ void __launch_bounds__(256) vtrans_k(const bf16* __restrict__ qkv,
                                                bf16* __restrict__ vT) {
  long i = (long)blockIdx.x * 256 + threadIdx.x;
  if (i >= (long)K_BATCH * K_HEADS * K_DHEAD * K_SEQ) return;
  int j = (int)(i % K_SEQ);
  long t = i / K_SEQ;
  int d = (int)(t % K_DHEAD); t /= K_DHEAD;
  int h = (int)(t % K_HEADS);
  int b = (int)(t / K_HEADS);
  vT[i] = qkv[((long)(b * K_SEQ + j)) * K_QKVN + 2 * K_DIM + h * K_DHEAD + d];
}

// ---------------- layernorm: f32 row -> bf16 row ----------------------------

__global__ void __launch_bounds__(256) layernorm_k(const float* __restrict__ x,
                                                   const float* __restrict__ g,
                                                   const float* __restrict__ b,
                                                   bf16* __restrict__ out) {
  int row = blockIdx.x;
  const float* xr = x + (size_t)row * K_DIM;
  float v[4]; float s = 0.f, q = 0.f;
  #pragma unroll
  for (int j = 0; j < 4; ++j) {
    v[j] = xr[threadIdx.x + 256 * j];
    s += v[j]; q += v[j] * v[j];
  }
  #pragma unroll
  for (int o = 16; o > 0; o >>= 1) {
    s += __shfl_xor(s, o, 32);
    q += __shfl_xor(q, o, 32);
  }
  __shared__ float rs[8], rq[8];
  int wave = threadIdx.x >> 5;
  if ((threadIdx.x & 31) == 0) { rs[wave] = s; rq[wave] = q; }
  __syncthreads();
  s = 0.f; q = 0.f;
  #pragma unroll
  for (int j = 0; j < 8; ++j) { s += rs[j]; q += rq[j]; }
  float mu  = s * (1.f / K_DIM);
  float var = q * (1.f / K_DIM) - mu * mu;
  float inv = rsqrtf(var + 1e-5f);
  bf16* orow = out + (size_t)row * K_DIM;
  #pragma unroll
  for (int j = 0; j < 4; ++j) {
    int idx = threadIdx.x + 256 * j;
    orow[idx] = (bf16)(((v[j] - mu) * inv) * g[idx] + b[idx]);
  }
}

// ---------------- generic WMMA GEMM with fused epilogue ---------------------
// C[M][N] = act(A[M][K] @ W + bias) + residual ; A bf16 row-major, W given as
// WT bf16 [N][K]; output f32 (Cf) or bf16 (Cb).
// Block: 256 thr = 8 waves (2 x 4), wave tile 32x32, block tile 64x128.

__global__ void __launch_bounds__(256) gemm_k(const bf16* __restrict__ A,
                                              const bf16* __restrict__ BT,
                                              int M, int N, int K,
                                              const float* __restrict__ bias,
                                              const float* __restrict__ residual,
                                              float* __restrict__ Cf,
                                              bf16* __restrict__ Cb,
                                              int gelu_flag) {
  int lane = threadIdx.x & 31;
  int wave = threadIdx.x >> 5;
  int wm = wave >> 2, wn = wave & 3;
  int mbase = blockIdx.y * 64 + wm * 32;
  int nbase = blockIdx.x * 128 + wn * 32;
  v8f c[2][2] = {};
  #pragma unroll 2
  for (int k0 = 0; k0 < K; k0 += 32) {
    v16bf a0 = load_frag_a(A, K, mbase,      k0);
    v16bf a1 = load_frag_a(A, K, mbase + 16, k0);
    v16bf b0 = load_frag_b(BT, K, k0, nbase);
    v16bf b1 = load_frag_b(BT, K, k0, nbase + 16);
    c[0][0] = wmma_bf16(a0, b0, c[0][0]);
    c[0][1] = wmma_bf16(a0, b1, c[0][1]);
    c[1][0] = wmma_bf16(a1, b0, c[1][0]);
    c[1][1] = wmma_bf16(a1, b1, c[1][1]);
  }
  int hs = lane >> 4, nlo = lane & 15;
  #pragma unroll
  for (int mi = 0; mi < 2; ++mi) {
    #pragma unroll
    for (int ni = 0; ni < 2; ++ni) {
      int col = nbase + ni * 16 + nlo;
      float bv = bias ? bias[col] : 0.f;
      #pragma unroll
      for (int r = 0; r < 8; ++r) {
        int row = mbase + mi * 16 + r + 8 * hs;
        float v = c[mi][ni][r] + bv;
        if (gelu_flag) v = 0.5f * v * (1.f + erff(v * 0.70710678118f));
        if (residual) v += residual[(size_t)row * N + col];
        if (Cf) Cf[(size_t)row * N + col] = v;
        else    Cb[(size_t)row * N + col] = (bf16)v;
      }
    }
  }
}

// ---------------- flash attention: wave = 16 query rows of one (b,h) --------
// S = Q Kt (WMMA), online softmax, O += P V (WMMA, V pre-transposed).

__global__ void __launch_bounds__(256) attn_k(const bf16* __restrict__ qkv,
                                              const bf16* __restrict__ vT,
                                              const unsigned char* __restrict__ mask,
                                              bf16* __restrict__ out) {
  __shared__ bf16 Pbuf[8][16 * 32];    // per-wave P tile (C-layout -> A-layout bounce)
  int lane = threadIdx.x & 31;
  int wave = threadIdx.x >> 5;
  int gw = blockIdx.x * 8 + wave;      // 4096 wave-jobs total
  int mblk = gw & 127;
  int h = (gw >> 7) & 15;
  int b = gw >> 11;

  const bf16* Q  = qkv + ((size_t)(b * K_SEQ + mblk * 16)) * K_QKVN + h * K_DHEAD;
  const bf16* Kp = qkv + ((size_t)b * K_SEQ) * K_QKVN + K_DIM + h * K_DHEAD;
  const bf16* Vt = vT + ((size_t)(b * K_HEADS + h)) * K_DHEAD * K_SEQ;
  const unsigned char* mb = mask + b * K_SEQ;
  bf16* Pw = &Pbuf[wave][0];

  v16bf aq0 = load_frag_a(Q, K_QKVN, 0, 0);    // Q dims 0..31
  v16bf aq1 = load_frag_a(Q, K_QKVN, 0, 32);   // Q dims 32..63
  v8f o0 = {}, o1 = {}, o2 = {}, o3 = {};
  float mrun[8], lrun[8];
  #pragma unroll
  for (int r = 0; r < 8; ++r) { mrun[r] = -1e30f; lrun[r] = 0.f; }
  int hs = lane >> 4, nlo = lane & 15;
  const float scale = 0.125f;                  // 1/sqrt(64)

  for (int j0 = 0; j0 < K_SEQ; j0 += 32) {
    // S tiles: keys j0..j0+15 (s0) and j0+16..j0+31 (s1); K rows double as B^T
    v16bf bk00 = load_frag_b(Kp, K_QKVN, 0,  j0);
    v16bf bk01 = load_frag_b(Kp, K_QKVN, 32, j0);
    v16bf bk10 = load_frag_b(Kp, K_QKVN, 0,  j0 + 16);
    v16bf bk11 = load_frag_b(Kp, K_QKVN, 32, j0 + 16);
    v8f s0 = {}, s1 = {};
    s0 = wmma_bf16(aq0, bk00, s0); s0 = wmma_bf16(aq1, bk01, s0);
    s1 = wmma_bf16(aq0, bk10, s1); s1 = wmma_bf16(aq1, bk11, s1);

    float msk0 = mb[j0 + nlo]      ? 0.f : -3e38f;
    float msk1 = mb[j0 + 16 + nlo] ? 0.f : -3e38f;
    float p0[8], p1[8], alpha[8];
    #pragma unroll
    for (int r = 0; r < 8; ++r) {
      float x0 = s0[r] * scale + msk0;
      float x1 = s1[r] * scale + msk1;
      float rm = half16_max(fmaxf(x0, x1));
      float mnew = fmaxf(mrun[r], rm);
      alpha[r] = __expf(mrun[r] - mnew);
      p0[r] = __expf(x0 - mnew);
      p1[r] = __expf(x1 - mnew);
      float rsum = half16_sum(p0[r] + p1[r]);
      lrun[r] = lrun[r] * alpha[r] + rsum;
      mrun[r] = mnew;
    }
    #pragma unroll
    for (int r = 0; r < 8; ++r) {
      o0[r] *= alpha[r]; o1[r] *= alpha[r];
      o2[r] *= alpha[r]; o3[r] *= alpha[r];
    }
    // C-layout P -> LDS (row-major 16x32), then reload as A fragment
    #pragma unroll
    for (int r = 0; r < 8; ++r) {
      int row = r + 8 * hs;
      Pw[row * 32 + nlo]      = (bf16)p0[r];
      Pw[row * 32 + 16 + nlo] = (bf16)p1[r];
    }
    __syncthreads();
    v16bf ap  = load_frag_a(Pw, 32, 0, 0);
    v16bf bv0 = load_frag_b(Vt, K_SEQ, j0, 0);
    v16bf bv1 = load_frag_b(Vt, K_SEQ, j0, 16);
    v16bf bv2 = load_frag_b(Vt, K_SEQ, j0, 32);
    v16bf bv3 = load_frag_b(Vt, K_SEQ, j0, 48);
    o0 = wmma_bf16(ap, bv0, o0);
    o1 = wmma_bf16(ap, bv1, o1);
    o2 = wmma_bf16(ap, bv2, o2);
    o3 = wmma_bf16(ap, bv3, o3);
    __syncthreads();
  }

  #pragma unroll
  for (int r = 0; r < 8; ++r) {
    float inv = 1.f / lrun[r];
    size_t rowg = (size_t)(b * K_SEQ + mblk * 16 + r + 8 * hs);
    bf16* op = out + rowg * K_DIM + h * K_DHEAD + nlo;
    op[0]  = (bf16)(o0[r] * inv);
    op[16] = (bf16)(o1[r] * inv);
    op[32] = (bf16)(o2[r] * inv);
    op[48] = (bf16)(o3[r] * inv);
  }
}

// ---------------------------------------------------------------------------

extern "C" void kernel_launch(void* const* d_in, const int* in_sizes, int n_in,
                              void* d_out, int out_size, void* d_ws, size_t ws_size,
                              hipStream_t stream) {
  (void)in_sizes; (void)n_in; (void)out_size; (void)ws_size;
  const float* x_in          = (const float*)d_in[0];
  const unsigned char* mask  = (const unsigned char*)d_in[1];
  const float* ln1_g = (const float*)d_in[2];
  const float* ln1_b = (const float*)d_in[3];
  const float* wqkv  = (const float*)d_in[4];
  const float* wo    = (const float*)d_in[5];
  const float* ln2_g = (const float*)d_in[6];
  const float* ln2_b = (const float*)d_in[7];
  const float* w1    = (const float*)d_in[8];
  const float* b1    = (const float*)d_in[9];
  const float* w2    = (const float*)d_in[10];
  const float* b2    = (const float*)d_in[11];
  float* xout = (float*)d_out;

  // workspace carve-up (~118 MB)
  char* ws = (char*)d_ws;
  size_t off = 0;
  auto carve = [&](size_t bytes) -> char* {
    char* p = ws + off;
    off += (bytes + 255) & ~(size_t)255;
    return p;
  };
  bf16* wqkvT = (bf16*)carve((size_t)K_QKVN * K_DIM * 2);
  bf16* woT   = (bf16*)carve((size_t)K_DIM  * K_DIM * 2);
  bf16* w1T   = (bf16*)carve((size_t)K_FF   * K_DIM * 2);
  bf16* w2T   = (bf16*)carve((size_t)K_DIM  * K_FF  * 2);
  bf16* xn    = (bf16*)carve((size_t)K_ROWS * K_DIM * 2);
  bf16* qkv   = (bf16*)carve((size_t)K_ROWS * K_QKVN * 2);
  bf16* vT    = (bf16*)carve((size_t)K_BATCH * K_HEADS * K_DHEAD * K_SEQ * 2);
  bf16* attnO = (bf16*)carve((size_t)K_ROWS * K_DIM * 2);
  bf16* h2    = (bf16*)carve((size_t)K_ROWS * K_DIM * 2);
  bf16* ff1o  = (bf16*)carve((size_t)K_ROWS * K_FF * 2);

  for (int l = 0; l < 4; ++l) {
    const float* wqkv_l = wqkv + (size_t)l * K_DIM * K_QKVN;
    const float* wo_l   = wo   + (size_t)l * K_DIM * K_DIM;
    const float* w1_l   = w1   + (size_t)l * K_DIM * K_FF;
    const float* w2_l   = w2   + (size_t)l * K_FF  * K_DIM;
    const float* resid  = (l == 0) ? x_in : xout;

    // repack weights f32 -> transposed bf16
    {
      long n = (long)K_DIM * K_QKVN;
      transpose_w_k<<<(n + 255) / 256, 256, 0, stream>>>(wqkv_l, wqkvT, K_DIM, K_QKVN);
      n = (long)K_DIM * K_DIM;
      transpose_w_k<<<(n + 255) / 256, 256, 0, stream>>>(wo_l, woT, K_DIM, K_DIM);
      n = (long)K_DIM * K_FF;
      transpose_w_k<<<(n + 255) / 256, 256, 0, stream>>>(w1_l, w1T, K_DIM, K_FF);
      transpose_w_k<<<(n + 255) / 256, 256, 0, stream>>>(w2_l, w2T, K_FF, K_DIM);
    }

    // LN1 -> xn (bf16)
    layernorm_k<<<K_ROWS, 256, 0, stream>>>(resid, ln1_g + l * K_DIM, ln1_b + l * K_DIM, xn);
    // QKV GEMM: [4096,1024] @ [1024,3072] -> bf16 [4096,3072]
    gemm_k<<<dim3(K_QKVN / 128, K_ROWS / 64), 256, 0, stream>>>(
        xn, wqkvT, K_ROWS, K_QKVN, K_DIM, nullptr, nullptr, nullptr, qkv, 0);
    // V transpose for P@V B-fragments
    {
      long n = (long)K_BATCH * K_HEADS * K_DHEAD * K_SEQ;
      vtrans_k<<<(n + 255) / 256, 256, 0, stream>>>(qkv, vT);
    }
    // flash attention: 4096 wave-jobs / 8 waves per block
    attn_k<<<512, 256, 0, stream>>>(qkv, vT, mask, attnO);
    // out projection + residual -> x (f32, d_out)
    gemm_k<<<dim3(K_DIM / 128, K_ROWS / 64), 256, 0, stream>>>(
        attnO, woT, K_ROWS, K_DIM, K_DIM, nullptr, resid, xout, nullptr, 0);
    // LN2 -> h2 (bf16)
    layernorm_k<<<K_ROWS, 256, 0, stream>>>(xout, ln2_g + l * K_DIM, ln2_b + l * K_DIM, h2);
    // FF1: gelu(h2 @ w1 + b1) -> bf16 [4096,4096]
    gemm_k<<<dim3(K_FF / 128, K_ROWS / 64), 256, 0, stream>>>(
        h2, w1T, K_ROWS, K_FF, K_DIM, b1 + (size_t)l * K_FF, nullptr, nullptr, ff1o, 1);
    // FF2: ff1o @ w2 + b2 + x -> x (f32, d_out)
    gemm_k<<<dim3(K_DIM / 128, K_ROWS / 64), 256, 0, stream>>>(
        ff1o, w2T, K_ROWS, K_DIM, K_FF, b2 + (size_t)l * K_DIM, xout, xout, nullptr, 0);
  }
}
